// FinalGNN_15247133900958
// MI455X (gfx1250) — matrix operand order repaired and checked by reading, not compiled, not run on hardware
//
#include <hip/hip_runtime.h>
#include <hip/hip_bf16.h>
#include <stdint.h>

#define NNODES 50000
#define NEDGES 800000
#define DINF   128
#define HDIM   512
#define GBATCH 64
#define EPSV   1e-5f

typedef __bf16 bf16_t;
typedef __attribute__((ext_vector_type(16))) __bf16 v16bf;
typedef __attribute__((ext_vector_type(8)))  float  v8f;

__device__ __forceinline__ unsigned short f2bf(float f) {
    unsigned u = __float_as_uint(f);
    u += 0x7FFFu + ((u >> 16) & 1u);   // round-to-nearest-even
    return (unsigned short)(u >> 16);
}

// ---------------------------------------------------------------- degree
__global__ void k_deg(const int* __restrict__ dst, float* __restrict__ deg) {
    int e = blockIdx.x * blockDim.x + threadIdx.x;
    if (e < NEDGES) atomicAdd(&deg[dst[e]], 1.0f);
}

// ----------------------------------------------- edge scatter: agg[dst]+=x[src]
__global__ void k_scatter(const float* __restrict__ x, const int* __restrict__ src,
                          const int* __restrict__ dst, float* __restrict__ agg,
                          int fquads /* F/4 */, int F) {
    int i = blockIdx.x * blockDim.x + threadIdx.x;
    int total = NEDGES * fquads;
    if (i >= total) return;
    int e  = i / fquads;
    int f4 = i - e * fquads;
    float4 v = *(const float4*)(x + (size_t)src[e] * F + f4 * 4);
    float* p = agg + (size_t)dst[e] * F + f4 * 4;
    atomicAdd(p + 0, v.x);
    atomicAdd(p + 1, v.y);
    atomicAdd(p + 2, v.z);
    atomicAdd(p + 3, v.w);
}

// -------------------------------------------- agg/deg -> bf16 (mean aggregate)
__global__ void k_divdeg_bf(const float* __restrict__ agg, const float* __restrict__ deg,
                            unsigned short* __restrict__ out, int fshift, int total) {
    int i = blockIdx.x * blockDim.x + threadIdx.x;
    if (i >= total) return;
    float d = deg[i >> fshift];
    d = d < 1.0f ? 1.0f : d;
    out[i] = f2bf(agg[i] / d);
}

// -------------------------------------------------------------- f32 -> bf16
__global__ void k_f2bf(const float* __restrict__ in, unsigned short* __restrict__ out, int total) {
    int i = blockIdx.x * blockDim.x + threadIdx.x;
    if (i < total) out[i] = f2bf(in[i]);
}

// ------------------------- weight transpose+convert: W[K][H] -> Wt[H][K] bf16
__global__ void k_wT(const float* __restrict__ W, unsigned short* __restrict__ Wt, int K) {
    int i = blockIdx.x * blockDim.x + threadIdx.x;
    if (i >= K * HDIM) return;
    int k = i / HDIM, n = i - k * HDIM;
    Wt[(size_t)n * K + k] = f2bf(W[i]);
}

// --------------------------------------------------------------- fused SAGE GEMM
// out[M][HDIM] = A1[M][K] @ B1 + A2[M][K] @ B2 + bias
// A* bf16 row-major; B*t bf16 column-major (Bt[n][k]).
// One block per 16-row band: the contiguous 16xK A panels (both streams) are
// staged in LDS once, then all 8 waves (one 16x64 output strip each) build
// their A fragments with ds_load_b128 instead of re-reading L2 8x.
__global__ __launch_bounds__(256)
void k_gemm_sage(const unsigned short* __restrict__ A1,
                 const unsigned short* __restrict__ A2,
                 const unsigned short* __restrict__ B1t,
                 const unsigned short* __restrict__ B2t,
                 const float* __restrict__ bias,
                 float* __restrict__ out, int M, int K) {
    __shared__ __align__(16) unsigned short lA1[16 * HDIM];   // max K = 512
    __shared__ __align__(16) unsigned short lA2[16 * HDIM];

    const int tid  = threadIdx.x;
    const int lane = tid & 31;
    const int row0 = blockIdx.x << 4;          // 16-row band
    const int col0 = (tid >> 5) << 6;          // wave -> 64-col strip

    // cooperative panel copy: 16 rows x K, rows contiguous in memory
    {
        const uint4* g1 = (const uint4*)(A1 + (size_t)row0 * K);
        const uint4* g2 = (const uint4*)(A2 + (size_t)row0 * K);
        uint4* s1 = (uint4*)lA1;
        uint4* s2 = (uint4*)lA2;
        const int nq = (16 * K) >> 3;          // 16-byte chunks
        for (int q = tid; q < nq; q += 256) { s1[q] = g1[q]; s2[q] = g2[q]; }
    }
    __syncthreads();

    // A frag (16-bit A 16x32): lanes 0-15 row M=lane hold K {0..15},
    // lanes 16-31 hold K {16..31}.
    const int akh = (lane >> 4) << 4;          // 0 or 16
    // B frag (32x16 from column-major): lane n = lane&15; lanes<16 K 0..15 else 16..31.
    const int bn  = lane & 15;
    const int bkh = (lane >> 4) << 4;          // 0 or 16

    v8f acc[4] = {};
    union Frag { v16bf v; uint4 q[2]; };

    const unsigned short* a1p = lA1 + (lane & 15) * K + akh;
    const unsigned short* a2p = lA2 + (lane & 15) * K + akh;

    for (int k0 = 0; k0 < K; k0 += 32) {
        Frag a1, a2;
        a1.q[0] = *(const uint4*)(a1p + k0);          // ds_load_b128
        a1.q[1] = *(const uint4*)(a1p + k0 + 8);
        a2.q[0] = *(const uint4*)(a2p + k0);
        a2.q[1] = *(const uint4*)(a2p + k0 + 8);
#pragma unroll
        for (int t = 0; t < 4; ++t) {
            const unsigned short* b1p = B1t + (size_t)(col0 + t * 16 + bn) * K + k0 + bkh;
            const unsigned short* b2p = B2t + (size_t)(col0 + t * 16 + bn) * K + k0 + bkh;
            if (t == 0) {                               // near-cache prefetch of next k-step B
                __builtin_prefetch(b1p + 32, 0, 3);
                __builtin_prefetch(b2p + 32, 0, 3);
            }
            Frag b1, b2;
            b1.q[0] = ((const uint4*)b1p)[0];
            b1.q[1] = ((const uint4*)b1p)[1];
            b2.q[0] = ((const uint4*)b2p)[0];
            b2.q[1] = ((const uint4*)b2p)[1];
            acc[t] = __builtin_amdgcn_wmma_f32_16x16x32_bf16(
                         false, a1.v, false, b1.v, (short)0, acc[t], false, false);
            acc[t] = __builtin_amdgcn_wmma_f32_16x16x32_bf16(
                         false, a2.v, false, b2.v, (short)0, acc[t], false, false);
        }
    }
    // D layout: VGPR r -> row = row0 + r + 8*(lane>=16), col = col0 + 16t + (lane&15)
    const int mrow = row0 + ((lane >> 4) << 3);
#pragma unroll
    for (int t = 0; t < 4; ++t) {
        int col = col0 + t * 16 + bn;
        float b = bias[col];
#pragma unroll
        for (int r = 0; r < 8; ++r)
            out[(size_t)(mrow + r) * HDIM + col] = acc[t][r] + b;
    }
}

// ------------------------------------------- per-column sum / sumsq reduction
__global__ void k_colstats(const float* __restrict__ h, float* __restrict__ s,
                           float* __restrict__ sq) {
    int c  = threadIdx.x;                 // 512 threads, one column each
    int r0 = blockIdx.x * 512;
    int r1 = r0 + 512; if (r1 > NNODES) r1 = NNODES;
    float a = 0.f, b = 0.f;
    for (int r = r0; r < r1; ++r) {
        float v = h[(size_t)r * HDIM + c];
        a += v; b += v * v;
    }
    atomicAdd(&s[c], a);
    atomicAdd(&sq[c], b);
}

// ------------------- BN (training stats) + ReLU + optional residual + bf16 copy
__global__ void k_bn_relu(const float* __restrict__ h, const float* __restrict__ s,
                          const float* __restrict__ sq, const float* __restrict__ gamma,
                          const float* __restrict__ beta, const float* __restrict__ res,
                          float* __restrict__ outF, unsigned short* __restrict__ outB) {
    int i = blockIdx.x * blockDim.x + threadIdx.x;
    if (i >= NNODES * HDIM) return;
    int c = i & (HDIM - 1);
    const float invN = 1.0f / (float)NNODES;
    float m   = s[c] * invN;
    float var = sq[c] * invN - m * m;
    float y   = gamma[c] * (h[i] - m) * rsqrtf(var + EPSV) + beta[c];
    y = fmaxf(y, 0.0f);
    if (res) y += res[i];
    outF[i] = y;
    outB[i] = f2bf(y);
}

// ---------------------------------------------------------------- pooling
__global__ void k_cnt(const int* __restrict__ batch, float* __restrict__ cnt) {
    int n = blockIdx.x * blockDim.x + threadIdx.x;
    if (n < NNODES) atomicAdd(&cnt[batch[n]], 1.0f);
}

__global__ void k_pool(const float* __restrict__ h, const int* __restrict__ batch,
                       float* __restrict__ sumP, unsigned int* __restrict__ maxP) {
    int i = blockIdx.x * blockDim.x + threadIdx.x;
    if (i >= NNODES * HDIM) return;
    int n = i >> 9;
    int c = i & (HDIM - 1);
    int g = batch[n];
    float v = h[i];                        // v >= 0 after ReLU chain
    atomicAdd(&sumP[g * HDIM + c], v);
    atomicMax(&maxP[g * HDIM + c], __float_as_uint(v));
}

// ------------------------------------------------------------------ head MLP
__global__ void k_fc1(const float* __restrict__ sumP, const unsigned int* __restrict__ maxP,
                      const float* __restrict__ cnt, const float* __restrict__ W,
                      const float* __restrict__ b, float* __restrict__ o1) {
    int i = blockIdx.x * blockDim.x + threadIdx.x;     // GBATCH*HDIM
    if (i >= GBATCH * HDIM) return;
    int g = i >> 9, j = i & (HDIM - 1);
    float c = cnt[g]; c = c < 1.0f ? 1.0f : c;
    float inv = 1.0f / c;
    float acc = b[j];
    for (int f = 0; f < HDIM; ++f) {
        float mp = sumP[g * HDIM + f] * inv;
        float xp = __uint_as_float(maxP[g * HDIM + f]);
        acc += mp * W[(size_t)f * HDIM + j] + xp * W[(size_t)(HDIM + f) * HDIM + j];
    }
    o1[i] = fmaxf(acc, 0.0f);
}

__global__ void k_bn_head(float* __restrict__ o1, const float* __restrict__ gh,
                          const float* __restrict__ bh) {
    int c = blockIdx.x * blockDim.x + threadIdx.x;
    if (c >= HDIM) return;
    float s = 0.f, sq = 0.f;
    for (int g = 0; g < GBATCH; ++g) { float v = o1[g * HDIM + c]; s += v; sq += v * v; }
    float m = s / (float)GBATCH;
    float var = sq / (float)GBATCH - m * m;
    float inv = rsqrtf(var + EPSV);
    for (int g = 0; g < GBATCH; ++g) {
        int idx = g * HDIM + c;
        o1[idx] = gh[c] * (o1[idx] - m) * inv + bh[c];
    }
}

__global__ void k_fc2(const float* __restrict__ in, const float* __restrict__ W,
                      const float* __restrict__ b, float* __restrict__ o2) {
    int i = blockIdx.x * blockDim.x + threadIdx.x;     // GBATCH*256
    if (i >= GBATCH * 256) return;
    int g = i >> 8, j = i & 255;
    float acc = b[j];
    for (int c = 0; c < HDIM; ++c) acc += in[g * HDIM + c] * W[c * 256 + j];
    o2[i] = fmaxf(acc, 0.0f);
}

__global__ void k_fc3(const float* __restrict__ in, const float* __restrict__ W,
                      const float* __restrict__ b, float* __restrict__ out) {
    int g = threadIdx.x;
    if (g >= GBATCH) return;
    float acc = b[0];
    for (int j = 0; j < 256; ++j) acc += in[g * 256 + j] * W[j];
    out[g] = acc;
}

// =============================================================== orchestration
extern "C" void kernel_launch(void* const* d_in, const int* in_sizes, int n_in,
                              void* d_out, int out_size, void* d_ws, size_t ws_size,
                              hipStream_t stream) {
    const float* x    = (const float*)d_in[0];
    const int*   ei   = (const int*)d_in[1];
    const int*   src  = ei;
    const int*   dst  = ei + NEDGES;
    const int*   batch= (const int*)d_in[2];
    const float* w1l = (const float*)d_in[3];  const float* b1l = (const float*)d_in[4];
    const float* w1r = (const float*)d_in[5];
    const float* w2l = (const float*)d_in[6];  const float* b2l = (const float*)d_in[7];
    const float* w2r = (const float*)d_in[8];
    const float* w3l = (const float*)d_in[9];  const float* b3l = (const float*)d_in[10];
    const float* w3r = (const float*)d_in[11];
    const float* g1  = (const float*)d_in[12]; const float* be1 = (const float*)d_in[13];
    const float* g2  = (const float*)d_in[14]; const float* be2 = (const float*)d_in[15];
    const float* g3  = (const float*)d_in[16]; const float* be3 = (const float*)d_in[17];
    const float* fc1w= (const float*)d_in[18]; const float* fc1b= (const float*)d_in[19];
    const float* gh  = (const float*)d_in[20]; const float* bh  = (const float*)d_in[21];
    const float* fc2w= (const float*)d_in[22]; const float* fc2b= (const float*)d_in[23];
    const float* fc3w= (const float*)d_in[24]; const float* fc3b= (const float*)d_in[25];

    char* ws = (char*)d_ws;
    auto carve = [&](size_t bytes) -> char* {
        char* p = ws; ws += (bytes + 255) & ~(size_t)255; return p;
    };
    float*          deg  = (float*)         carve((size_t)NNODES * 4);
    float*          aggF = (float*)         carve((size_t)NNODES * HDIM * 4);
    unsigned short* aggB = (unsigned short*)carve((size_t)NNODES * HDIM * 2);
    unsigned short* xB   = (unsigned short*)carve((size_t)NNODES * HDIM * 2);
    float*          hcur = (float*)         carve((size_t)NNODES * HDIM * 4);
    float*          hres = (float*)         carve((size_t)NNODES * HDIM * 4);
    unsigned short* wlt  = (unsigned short*)carve((size_t)HDIM * HDIM * 2);
    unsigned short* wrt  = (unsigned short*)carve((size_t)HDIM * HDIM * 2);
    float*          cs   = (float*)         carve(HDIM * 4);
    float*          csq  = (float*)         carve(HDIM * 4);
    float*          sumP = (float*)         carve((size_t)GBATCH * HDIM * 4);
    unsigned int*   maxP = (unsigned int*)  carve((size_t)GBATCH * HDIM * 4);
    float*          cnt  = (float*)         carve(GBATCH * 4);
    float*          o1   = (float*)         carve((size_t)GBATCH * HDIM * 4);
    float*          o2   = (float*)         carve((size_t)GBATCH * 256 * 4);

    const int T = 256;
    const int NH = NNODES * HDIM;                 // 25.6M
    const int gemmBlocks = NNODES / 16;           // one block per 16-row band (3125)

    // ---- degree (shared by all layers)
    hipMemsetAsync(deg, 0, (size_t)NNODES * 4, stream);
    k_deg<<<(NEDGES + T - 1) / T, T, 0, stream>>>(dst, deg);

    // ================= layer 1 (K = 128) =================
    hipMemsetAsync(aggF, 0, (size_t)NNODES * DINF * 4, stream);
    {
        int total = NEDGES * (DINF / 4);
        k_scatter<<<(total + T - 1) / T, T, 0, stream>>>(x, src, dst, aggF, DINF / 4, DINF);
    }
    k_divdeg_bf<<<(NNODES * DINF + T - 1) / T, T, 0, stream>>>(aggF, deg, aggB, 7, NNODES * DINF);
    k_f2bf<<<(NNODES * DINF + T - 1) / T, T, 0, stream>>>(x, xB, NNODES * DINF);
    k_wT<<<(DINF * HDIM + T - 1) / T, T, 0, stream>>>(w1l, wlt, DINF);
    k_wT<<<(DINF * HDIM + T - 1) / T, T, 0, stream>>>(w1r, wrt, DINF);
    k_gemm_sage<<<gemmBlocks, T, 0, stream>>>(aggB, xB, wlt, wrt, b1l, hcur, NNODES, DINF);
    hipMemsetAsync(cs, 0, HDIM * 4, stream);
    hipMemsetAsync(csq, 0, HDIM * 4, stream);
    k_colstats<<<(NNODES + 511) / 512, 512, 0, stream>>>(hcur, cs, csq);
    k_bn_relu<<<(NH + T - 1) / T, T, 0, stream>>>(hcur, cs, csq, g1, be1, nullptr, hres, xB);

    // ================= layers 2 & 3 (K = 512) =================
    const float* wl[2] = { w2l, w3l };
    const float* wr[2] = { w2r, w3r };
    const float* bl[2] = { b2l, b3l };
    const float* gm[2] = { g2, g3 };
    const float* bt[2] = { be2, be3 };
    for (int L = 0; L < 2; ++L) {
        hipMemsetAsync(aggF, 0, (size_t)NNODES * HDIM * 4, stream);
        int total = NEDGES * (HDIM / 4);
        k_scatter<<<(total + T - 1) / T, T, 0, stream>>>(hres, src, dst, aggF, HDIM / 4, HDIM);
        k_divdeg_bf<<<(NH + T - 1) / T, T, 0, stream>>>(aggF, deg, aggB, 9, NH);
        k_wT<<<(HDIM * HDIM + T - 1) / T, T, 0, stream>>>(wl[L], wlt, HDIM);
        k_wT<<<(HDIM * HDIM + T - 1) / T, T, 0, stream>>>(wr[L], wrt, HDIM);
        k_gemm_sage<<<gemmBlocks, T, 0, stream>>>(aggB, xB, wlt, wrt, bl[L], hcur, NNODES, HDIM);
        hipMemsetAsync(cs, 0, HDIM * 4, stream);
        hipMemsetAsync(csq, 0, HDIM * 4, stream);
        k_colstats<<<(NNODES + 511) / 512, 512, 0, stream>>>(hcur, cs, csq);
        k_bn_relu<<<(NH + T - 1) / T, T, 0, stream>>>(hcur, cs, csq, gm[L], bt[L], hres, hres, xB);
    }

    // ================= pooling + head =================
    hipMemsetAsync(sumP, 0, (size_t)GBATCH * HDIM * 4, stream);
    hipMemsetAsync(maxP, 0, (size_t)GBATCH * HDIM * 4, stream);   // values >= 0
    hipMemsetAsync(cnt, 0, GBATCH * 4, stream);
    k_cnt<<<(NNODES + T - 1) / T, T, 0, stream>>>(batch, cnt);
    k_pool<<<(NH + T - 1) / T, T, 0, stream>>>(hres, batch, sumP, maxP);

    k_fc1<<<(GBATCH * HDIM + T - 1) / T, T, 0, stream>>>(sumP, maxP, cnt, fc1w, fc1b, o1);
    k_bn_head<<<(HDIM + T - 1) / T, T, 0, stream>>>(o1, gh, bh);
    k_fc2<<<(GBATCH * 256 + T - 1) / T, T, 0, stream>>>(o1, fc2w, fc2b, o2);
    k_fc3<<<1, GBATCH, 0, stream>>>(o2, fc3w, fc3b, (float*)d_out);
}
